// Decoder_RNN_85985245266063
// MI455X (gfx1250) — matrix operand order repaired and checked by reading, not compiled, not run on hardware
//
#include <hip/hip_runtime.h>

// ---------------------------------------------------------------------------
// Types for CDNA5 WMMA (gfx1250, wave32)
// ---------------------------------------------------------------------------
typedef __attribute__((ext_vector_type(16))) __bf16 v16bf;
typedef __attribute__((ext_vector_type(8)))  float  v8f;

__device__ __forceinline__ __bf16 f2bf(float f) {
    unsigned u = __builtin_bit_cast(unsigned, f);
    unsigned r = u + 0x7FFFu + ((u >> 16) & 1u);   // round-to-nearest-even
    unsigned short h = (unsigned short)(r >> 16);
    return __builtin_bit_cast(__bf16, h);
}
__device__ __forceinline__ float bf2f(__bf16 b) {
    unsigned short h = __builtin_bit_cast(unsigned short, b);
    unsigned u = ((unsigned)h) << 16;
    return __builtin_bit_cast(float, u);
}
__device__ __forceinline__ float bfu_lo(unsigned w) {   // low 16 bits as bf16 -> f32
    return __builtin_bit_cast(float, w << 16);
}
__device__ __forceinline__ float bfu_hi(unsigned w) {   // high 16 bits as bf16 -> f32
    return __builtin_bit_cast(float, w & 0xffff0000u);
}
__device__ __forceinline__ float fast_sigmoid(float x) {
    return 1.f / (1.f + __expf(-x));
}
__device__ __forceinline__ float fast_tanh(float x) {
    float e = __expf(-2.f * fabsf(x));
    float r = (1.f - e) / (1.f + e);
    return copysignf(r, x);
}

// Dimensions (fixed by the reference problem)
#define Bz   128
#define Lz   1024
#define KVz  256
#define Ez   256
#define Hz   512
#define Vz   1000
#define MHz  1024
#define Tz   256

// ---------------------------------------------------------------------------
// Generic 16x16 WMMA tile: C += A[MxK](row-major bf16, lda) * W[NxK]^T
// A frag (16x32 bf16): lane row = lane&15; 16 contiguous K starting at 16h, h=lane>>4
// B frag from W[N,K] row-major: lane holds column n0+(lane&15), 16 contiguous K
// D: lane col = lane&15; vgpr v holds m = v + 8*(lane>>4)
// ---------------------------------------------------------------------------
__device__ __forceinline__ v8f wmma_tile(const __bf16* __restrict__ A, int lda,
                                         const __bf16* __restrict__ W, int ldw,
                                         int m0, int n0, int K, int lane) {
    v8f acc = {};
    const int row  = lane & 15;
    const int half = lane >> 4;
    const __bf16* Ap = A + (size_t)(m0 + row) * lda + half * 16;
    const __bf16* Bp = W + (size_t)(n0 + row) * ldw + half * 16;
    for (int k0 = 0; k0 < K; k0 += 32) {
        v16bf a, b;
#pragma unroll
        for (int i = 0; i < 16; ++i) a[i] = Ap[k0 + i];
#pragma unroll
        for (int i = 0; i < 16; ++i) b[i] = Bp[k0 + i];
        __builtin_prefetch(Ap + k0 + 32, 0, 0);
        __builtin_prefetch(Bp + k0 + 32, 0, 0);
        acc = __builtin_amdgcn_wmma_f32_16x16x32_bf16(false, a, false, b,
                                                      (short)0, acc, false, false);
    }
    return acc;
}

// One 256-wide K segment of the LSTM GEMM: acc[g] += A_seg x W[:, kbase..kbase+255]^T
__device__ __forceinline__ void lstm_seg(const __bf16* __restrict__ seg, int ld, int kbase,
                                         const __bf16* __restrict__ W, int m0, int ncol0,
                                         int row, int half, v8f acc[4]) {
    const __bf16* Ap = seg + (size_t)(m0 + row) * ld + half * 16;
    for (int k0 = 0; k0 < 256; k0 += 32) {
        v16bf a;
#pragma unroll
        for (int i = 0; i < 16; ++i) a[i] = Ap[k0 + i];
        __builtin_prefetch(Ap + k0 + 32, 0, 0);
#pragma unroll
        for (int g = 0; g < 4; ++g) {
            const __bf16* Bp = W + (size_t)(g * 512 + ncol0 + row) * 1024 + kbase + k0 + half * 16;
            v16bf b;
#pragma unroll
            for (int i = 0; i < 16; ++i) b[i] = Bp[i];
            acc[g] = __builtin_amdgcn_wmma_f32_16x16x32_bf16(false, a, false, b,
                                                             (short)0, acc[g], false, false);
        }
    }
}

// ---------------------------------------------------------------------------
// Setup kernels (run once per launch)
// ---------------------------------------------------------------------------
__global__ void convert_kv_kernel(const float* __restrict__ key, const float* __restrict__ val,
                                  __bf16* __restrict__ kbf, __bf16* __restrict__ vbf, int n) {
    int i = blockIdx.x * 256 + threadIdx.x;
    if (i < n) { kbf[i] = f2bf(key[i]); vbf[i] = f2bf(val[i]); }
}

__global__ void build_wcat_kernel(const float* __restrict__ W0ih, const float* __restrict__ W0hh,
                                  const float* __restrict__ Wrih, const float* __restrict__ Wrhh,
                                  __bf16* __restrict__ Wcat) {
    int i = blockIdx.x * 256 + threadIdx.x;           // < 3*2048*1024
    int layer = i >> 21;
    int rem   = i & ((1 << 21) - 1);
    int r = rem >> 10, k = rem & 1023;
    float v;
    if (layer == 0) v = (k < 512) ? W0ih[r * 512 + k] : W0hh[r * 512 + (k - 512)];
    else {
        int l = layer - 1;
        v = (k < 512) ? Wrih[((size_t)l * 2048 + r) * 512 + k]
                      : Wrhh[((size_t)l * 2048 + r) * 512 + (k - 512)];
    }
    Wcat[i] = f2bf(v);
}

__global__ void build_small_kernel(const float* __restrict__ fcW, const float* __restrict__ W1,
                                   const float* __restrict__ W2,
                                   __bf16* __restrict__ fcWb, __bf16* __restrict__ W1b,
                                   __bf16* __restrict__ W2b) {
    int i = blockIdx.x * 256 + threadIdx.x;           // grid covers 1,048,576
    if (i < 256 * 512)  fcWb[i] = f2bf(fcW[i]);
    if (i < 1024 * 512) W1b[i]  = f2bf(W1[i]);
    if (i < 1024 * 1024) {
        int r = i >> 10;
        W2b[i] = (r < Vz) ? f2bf(W2[(size_t)r * 1024 + (i & 1023)]) : f2bf(0.f);
    }
}

__global__ void embed_kernel(const float* __restrict__ embW, const int* __restrict__ labels,
                             const int* __restrict__ sos, __bf16* __restrict__ xs) {
    int i = blockIdx.x * 256 + threadIdx.x;           // T*B*E = 8,388,608
    int j = i & 255, b = (i >> 8) & 127, t = i >> 15;
    int tok = (t == 0) ? sos[0] : labels[b * 256 + (t - 1)];
    xs[i] = f2bf(embW[(size_t)tok * Ez + j]);
}

__global__ void init_state_kernel(__bf16* __restrict__ hbf, float* __restrict__ cbuf,
                                  __bf16* __restrict__ zin) {
    int i = blockIdx.x * 256 + threadIdx.x;           // grid covers 393216
    if (i < 2 * 3 * Bz * Hz) hbf[i] = f2bf(0.f);      // both ping-pong parities
    if (i < 3 * Bz * Hz)     cbuf[i] = 0.f;
    if (i < Bz * 512)        zin[i] = f2bf(0.f);
}

// ---------------------------------------------------------------------------
// LSTM layer: gates = [seg0|seg1|seg2|seg3] * Wcat[2048,1024]^T + b, fused pointwise.
// Each wave owns one 16x16 tile for all 4 gates (cols spaced H apart) -> i,f,g,o
// in-register. hout must NOT alias any seg (ping-pong); c updates are element-owned.
// ---------------------------------------------------------------------------
__global__ __launch_bounds__(256)
void lstm_layer_kernel(const __bf16* __restrict__ s0, int ld0,
                       const __bf16* __restrict__ s1, int ld1,
                       const __bf16* __restrict__ s2, int ld2,
                       const __bf16* __restrict__ s3, int ld3,
                       const __bf16* __restrict__ W, const float* __restrict__ bias,
                       float* __restrict__ c, __bf16* __restrict__ hout) {
    const int lane  = threadIdx.x & 31;
    const int wave  = threadIdx.x >> 5;                   // 0..7
    const int m0    = blockIdx.x * 16;                    // blockIdx.x: 0..7
    const int ncol0 = (blockIdx.y * 8 + wave) * 16;       // 0..511
    const int row = lane & 15, half = lane >> 4;

    v8f acc[4] = {};
    lstm_seg(s0, ld0,   0, W, m0, ncol0, row, half, acc);
    lstm_seg(s1, ld1, 256, W, m0, ncol0, row, half, acc);
    lstm_seg(s2, ld2, 512, W, m0, ncol0, row, half, acc);
    lstm_seg(s3, ld3, 768, W, m0, ncol0, row, half, acc);

    const int col = ncol0 + (lane & 15);
    const float bi = bias[col], bff = bias[512 + col], bg = bias[1024 + col], bo = bias[1536 + col];
#pragma unroll
    for (int v = 0; v < 8; ++v) {
        int m = m0 + v + half * 8;
        float gi = acc[0][v] + bi, gf = acc[1][v] + bff;
        float gg = acc[2][v] + bg, go = acc[3][v] + bo;
        float cn = fast_sigmoid(gf) * c[m * 512 + col] + fast_sigmoid(gi) * fast_tanh(gg);
        c[m * 512 + col] = cn;
        hout[m * 512 + col] = f2bf(fast_sigmoid(go) * fast_tanh(cn));
    }
}

// query = h2 * fcW^T + fc_b -> fp32 (for attention) + bf16 into zin[:,0:256]
__global__ __launch_bounds__(256)
void query_kernel(const __bf16* __restrict__ h2, const __bf16* __restrict__ fcWb,
                  const float* __restrict__ fcb, float* __restrict__ qf,
                  __bf16* __restrict__ zin) {
    int lane = threadIdx.x & 31;
    int widx = blockIdx.x * 8 + (threadIdx.x >> 5);       // 0..127 -> 8 x 16 tiles
    int m0 = (widx >> 4) * 16, n0 = (widx & 15) * 16;
    v8f acc = wmma_tile(h2, 512, fcWb, 512, m0, n0, 512, lane);
    int col = n0 + (lane & 15), half = lane >> 4;
    float b = fcb[col];
#pragma unroll
    for (int v = 0; v < 8; ++v) {
        int m = m0 + v + half * 8;
        float x = acc[v] + b;
        qf[m * 256 + col] = x;
        zin[m * 512 + col] = f2bf(x);
    }
}

// Fused attention: energy (wave-per-row b128 dots) -> softmax(energy*mask) -> att
// -> ctx = att . value (wave-parallel over L, LDS-reduced) into zin[:,256:512].
// One block (8 waves) per batch row.
// tp=1: att written step-major [T][B][L] (coalesced, transposed later)
// tp=0: att written directly to [B][L][T]
__global__ __launch_bounds__(256)
void attention_kernel(const __bf16* __restrict__ kbf, const __bf16* __restrict__ vbf,
                      const float* __restrict__ qf, const int* __restrict__ seqlens,
                      float* __restrict__ attdst, __bf16* __restrict__ zin, int t, int tp) {
    __shared__ float e[1024];
    __shared__ float red[256];
    __shared__ float part[8][256];
    const int b = blockIdx.x, tid = threadIdx.x;
    const int lane = tid & 31, wave = tid >> 5;
    const int sl = seqlens[b];

    // each lane owns 8 fixed query channels: hoist them into registers
    float qv[8];
#pragma unroll
    for (int i = 0; i < 8; ++i) qv[i] = qf[b * 256 + lane * 8 + i];

    // energies: one wave per key row; 16B vector loads; multiplicative mask
    const __bf16* kb = kbf + (size_t)b * Lz * KVz;
    for (int l = wave; l < 1024; l += 8) {
        uint4 d = *(const uint4*)(kb + (size_t)l * 256 + lane * 8);
        float s = bfu_lo(d.x) * qv[0] + bfu_hi(d.x) * qv[1]
                + bfu_lo(d.y) * qv[2] + bfu_hi(d.y) * qv[3]
                + bfu_lo(d.z) * qv[4] + bfu_hi(d.z) * qv[5]
                + bfu_lo(d.w) * qv[6] + bfu_hi(d.w) * qv[7];
#pragma unroll
        for (int off = 16; off > 0; off >>= 1) s += __shfl_xor(s, off);
        if (lane == 0) e[l] = (l < sl) ? s : 0.f;
    }
    __syncthreads();

    float m = -3.4e38f;
    for (int j = tid; j < 1024; j += 256) m = fmaxf(m, e[j]);
    red[tid] = m; __syncthreads();
    for (int s2 = 128; s2 > 0; s2 >>= 1) { if (tid < s2) red[tid] = fmaxf(red[tid], red[tid + s2]); __syncthreads(); }
    float mx = red[0]; __syncthreads();
    float sum = 0.f;
    for (int j = tid; j < 1024; j += 256) { float ex = __expf(e[j] - mx); e[j] = ex; sum += ex; }
    red[tid] = sum; __syncthreads();
    for (int s2 = 128; s2 > 0; s2 >>= 1) { if (tid < s2) red[tid] += red[tid + s2]; __syncthreads(); }
    float inv = 1.f / red[0]; __syncthreads();
    for (int j = tid; j < 1024; j += 256) {
        float a = e[j] * inv; e[j] = a;
        size_t idx = tp ? (((size_t)t * Bz + b) * Lz + j)
                        : (((size_t)(b * Lz + j)) * Tz + t);
        attdst[idx] = a;
    }
    __syncthreads();

    // ctx: each wave accumulates 128 rows into 8 channels/lane, LDS-reduce waves
    float acc[8] = {};
    const __bf16* vb = vbf + (size_t)b * Lz * KVz;
    for (int l = wave; l < 1024; l += 8) {
        float a = e[l];
        uint4 d = *(const uint4*)(vb + (size_t)l * 256 + lane * 8);
        acc[0] += a * bfu_lo(d.x); acc[1] += a * bfu_hi(d.x);
        acc[2] += a * bfu_lo(d.y); acc[3] += a * bfu_hi(d.y);
        acc[4] += a * bfu_lo(d.z); acc[5] += a * bfu_hi(d.z);
        acc[6] += a * bfu_lo(d.w); acc[7] += a * bfu_hi(d.w);
    }
#pragma unroll
    for (int j = 0; j < 8; ++j) part[wave][lane * 8 + j] = acc[j];
    __syncthreads();
    float ctx = 0.f;
#pragma unroll
    for (int g = 0; g < 8; ++g) ctx += part[g][tid];
    zin[b * 512 + 256 + tid] = f2bf(ctx);                 // ctx for next step's layer-0 input
}

// Final transpose: att [T][B][L] -> attentions [B][L][T], LDS-tiled
__global__ __launch_bounds__(256)
void att_transpose_kernel(const float* __restrict__ in, float* __restrict__ out) {
    __shared__ float tile[32][33];
    const int b = blockIdx.z;
    const int l0 = blockIdx.x * 32, t0 = blockIdx.y * 32;
    const int tx = threadIdx.x, ty = threadIdx.y;         // 32 x 8
#pragma unroll
    for (int r = 0; r < 4; ++r) {
        int tt = t0 + ty + r * 8;
        tile[ty + r * 8][tx] = in[((size_t)tt * Bz + b) * Lz + l0 + tx];
    }
    __syncthreads();
#pragma unroll
    for (int r = 0; r < 4; ++r) {
        int ll = l0 + ty + r * 8;
        out[((size_t)b * Lz + ll) * Tz + t0 + tx] = tile[tx][ty + r * 8];
    }
}

// hmid = leaky(zin * W1^T + b1), bf16
__global__ __launch_bounds__(256)
void mlp1_kernel(const __bf16* __restrict__ zin, const __bf16* __restrict__ W1b,
                 const float* __restrict__ b1, __bf16* __restrict__ hmid) {
    int lane = threadIdx.x & 31;
    int widx = blockIdx.x * 8 + (threadIdx.x >> 5);       // 0..511 -> 8 x 64 tiles
    int m0 = (widx >> 6) * 16, n0 = (widx & 63) * 16;
    v8f acc = wmma_tile(zin, 512, W1b, 512, m0, n0, 512, lane);
    int col = n0 + (lane & 15), half = lane >> 4;
    float b = b1[col];
#pragma unroll
    for (int v = 0; v < 8; ++v) {
        float x = acc[v] + b;
        x = (x >= 0.f) ? x : 0.9f * x;                    // LeakyReLU(0.9)
        hmid[(size_t)(m0 + v + half * 8) * 1024 + col] = f2bf(x);
    }
}

// logits -> y_hat[b, t, :]  (N padded 1000 -> 1024 with zero rows)
__global__ __launch_bounds__(256)
void mlp2_kernel(const __bf16* __restrict__ hmid, const __bf16* __restrict__ W2b,
                 const float* __restrict__ b2, float* __restrict__ yhat, int t) {
    int lane = threadIdx.x & 31;
    int widx = blockIdx.x * 8 + (threadIdx.x >> 5);       // 0..511 -> 8 x 64 tiles
    int m0 = (widx >> 6) * 16, n0 = (widx & 63) * 16;
    v8f acc = wmma_tile(hmid, 1024, W2b, 1024, m0, n0, 1024, lane);
    int col = n0 + (lane & 15), half = lane >> 4;
    if (col < Vz) {
        float b = b2[col];
#pragma unroll
        for (int v = 0; v < 8; ++v) {
            int m = m0 + v + half * 8;
            yhat[((size_t)m * Tz + t) * Vz + col] = acc[v] + b;
        }
    }
}

__global__ void argmax_kernel(const float* __restrict__ yhat, float* __restrict__ label) {
    int p = blockIdx.x * 256 + threadIdx.x;               // B*T = 32768
    const float* row = yhat + (size_t)p * Vz;
    float best = row[0]; int bi = 0;
    for (int v = 1; v < Vz; ++v) { float x = row[v]; if (x > best) { best = x; bi = v; } }
    label[p] = (float)bi;
}

// ---------------------------------------------------------------------------
// Workspace layout (bytes). Base ~169 MB (key/value bf16 stay L2-resident:
// 134 MB < 192 MB global L2). Optional +134 MB step-major attention staging.
// ---------------------------------------------------------------------------
static constexpr size_t OFF_KBF  = 0;
static constexpr size_t OFF_VBF  = OFF_KBF  + 67108864ull;
static constexpr size_t OFF_WCAT = OFF_VBF  + 67108864ull;      // 12,582,912
static constexpr size_t OFF_FCW  = OFF_WCAT + 12582912ull;      //    262,144
static constexpr size_t OFF_W1   = OFF_FCW  + 262144ull;        //  1,048,576
static constexpr size_t OFF_W2   = OFF_W1   + 1048576ull;       //  2,097,152
static constexpr size_t OFF_XS   = OFF_W2   + 2097152ull;       // 16,777,216
static constexpr size_t OFF_HBF  = OFF_XS   + 16777216ull;      //    786,432 (2 x 3 x B x H bf16)
static constexpr size_t OFF_CB   = OFF_HBF  + 786432ull;        //    786,432
static constexpr size_t OFF_ZIN  = OFF_CB   + 786432ull;        //    131,072
static constexpr size_t OFF_HMID = OFF_ZIN  + 131072ull;        //    262,144
static constexpr size_t OFF_QF   = OFF_HMID + 262144ull;        //    131,072
static constexpr size_t OFF_ATT  = OFF_QF   + 131072ull;        // optional 134,217,728
static constexpr size_t NEED_ATT = OFF_ATT  + 134217728ull;

extern "C" void kernel_launch(void* const* d_in, const int* in_sizes, int n_in,
                              void* d_out, int out_size, void* d_ws, size_t ws_size,
                              hipStream_t stream) {
    const float* key       = (const float*)d_in[0];
    const float* value     = (const float*)d_in[1];
    const int*   labels    = (const int*)d_in[2];
    const int*   seqlens   = (const int*)d_in[3];
    const int*   sos       = (const int*)d_in[4];
    const float* embed_W   = (const float*)d_in[5];
    const float* lstm0_Wih = (const float*)d_in[6];
    const float* lstm0_Whh = (const float*)d_in[7];
    const float* lstm0_b   = (const float*)d_in[8];
    const float* lstmr_Wih = (const float*)d_in[9];
    const float* lstmr_Whh = (const float*)d_in[10];
    const float* lstmr_b   = (const float*)d_in[11];
    const float* fc_W      = (const float*)d_in[12];
    const float* fc_b      = (const float*)d_in[13];
    const float* mlp_W1    = (const float*)d_in[14];
    const float* mlp_b1    = (const float*)d_in[15];
    const float* mlp_W2    = (const float*)d_in[16];
    const float* mlp_b2    = (const float*)d_in[17];

    char* ws = (char*)d_ws;
    __bf16* kbf   = (__bf16*)(ws + OFF_KBF);
    __bf16* vbf   = (__bf16*)(ws + OFF_VBF);
    __bf16* Wcat  = (__bf16*)(ws + OFF_WCAT);   // 3 x [2048 x 1024]
    __bf16* fcWb  = (__bf16*)(ws + OFF_FCW);
    __bf16* W1b   = (__bf16*)(ws + OFF_W1);
    __bf16* W2b   = (__bf16*)(ws + OFF_W2);
    __bf16* xs    = (__bf16*)(ws + OFF_XS);     // [T][B][E]
    __bf16* hbf   = (__bf16*)(ws + OFF_HBF);    // [2 parities][3 layers][B][H]
    float*  cbuf  = (float*) (ws + OFF_CB);     // [3][B][H]
    __bf16* zin   = (__bf16*)(ws + OFF_ZIN);    // [B][512] = [query | ctx]
    __bf16* hmid  = (__bf16*)(ws + OFF_HMID);   // [B][1024]
    float*  qf    = (float*) (ws + OFF_QF);     // [B][256]
    float*  attws = (float*) (ws + OFF_ATT);    // [T][B][L] (optional staging)

    float* yhat   = (float*)d_out;                               // [B][T][V]
    float* label  = yhat + (size_t)Bz * Tz * Vz;                 // [B][T]
    float* attout = label + (size_t)Bz * Tz;                     // [B][L][T]

    const int useT = (ws_size >= NEED_ATT) ? 1 : 0;              // deterministic per harness

    // ---- one-time conversions (per launch; deterministic) ----
    convert_kv_kernel<<<131072, 256, 0, stream>>>(key, value, kbf, vbf, Bz * Lz * KVz);
    build_wcat_kernel<<<24576, 256, 0, stream>>>(lstm0_Wih, lstm0_Whh, lstmr_Wih, lstmr_Whh, Wcat);
    build_small_kernel<<<4096, 256, 0, stream>>>(fc_W, mlp_W1, mlp_W2, fcWb, W1b, W2b);
    embed_kernel<<<32768, 256, 0, stream>>>(embed_W, labels, sos, xs);
    init_state_kernel<<<1536, 256, 0, stream>>>(hbf, cbuf, zin);

    const size_t Wstride = 2048ull * 1024ull;    // elements per layer weight block
    const size_t Hs      = (size_t)Bz * Hz;      // 65536 elements per h/c plane
    float* attdst = useT ? attws : attout;

    for (int t = 0; t < Tz; ++t) {
        const int p = t & 1, pn = p ^ 1;
        __bf16* h0o = hbf + (size_t)p  * 3 * Hs;     // old parity planes
        __bf16* h1o = h0o + Hs;
        __bf16* h2o = h1o + Hs;
        __bf16* h0n = hbf + (size_t)pn * 3 * Hs;     // new parity planes
        __bf16* h1n = h0n + Hs;
        __bf16* h2n = h1n + Hs;
        const __bf16* xs_t = xs + (size_t)t * Bz * Ez;

        // layer 0: A = [x_emb | ctx | h0_old]  (ctx lives at zin[:,256:512])
        lstm_layer_kernel<<<dim3(8, 4), 256, 0, stream>>>(
            xs_t, 256, zin + 256, 512, h0o, 512, h0o + 256, 512,
            Wcat, lstm0_b, cbuf, h0n);
        // layer 1: A = [h0_new | h1_old]
        lstm_layer_kernel<<<dim3(8, 4), 256, 0, stream>>>(
            h0n, 512, h0n + 256, 512, h1o, 512, h1o + 256, 512,
            Wcat + Wstride, lstmr_b, cbuf + Hs, h1n);
        // layer 2: A = [h1_new | h2_old]
        lstm_layer_kernel<<<dim3(8, 4), 256, 0, stream>>>(
            h1n, 512, h1n + 256, 512, h2o, 512, h2o + 256, 512,
            Wcat + 2 * Wstride, lstmr_b + 2048, cbuf + 2 * Hs, h2n);

        query_kernel<<<16, 256, 0, stream>>>(h2n, fcWb, fc_b, qf, zin);
        attention_kernel<<<128, 256, 0, stream>>>(kbf, vbf, qf, seqlens, attdst, zin, t, useT);
        mlp1_kernel<<<64, 256, 0, stream>>>(zin, W1b, mlp_b1, hmid);
        mlp2_kernel<<<64, 256, 0, stream>>>(hmid, W2b, mlp_b2, yhat, t);
    }

    if (useT)
        att_transpose_kernel<<<dim3(32, 8, 128), dim3(32, 8), 0, stream>>>(attws, attout);
    argmax_kernel<<<128, 256, 0, stream>>>(yhat, label);
}
